// My_GAT_55241869361816
// MI455X (gfx1250) — matrix operand order repaired and checked by reading, not compiled, not run on hardware
//
#include <hip/hip_runtime.h>
#include <stdint.h>

// ---------------------------------------------------------------------------
// GAT forward for MI455X (gfx1250, wave32).
// GEMMs via v_wmma_f32_16x16x32_bf16 with pre-converted bf16 operands
// (no f32->bf16 cvt in the inner loop). Edge softmax-aggregation via
// hardware f32 atomics on an L2-resident (10 MB) working set.
// ---------------------------------------------------------------------------

#define N_NODES 20000
#define N_EDGES 640000
#define IN_DIM  24
#define HDIM    128
#define OUTD    24
#define MAPD    512

typedef __bf16 bf16;
typedef bf16  v16bf __attribute__((ext_vector_type(16)));
typedef bf16  v8bf  __attribute__((ext_vector_type(8)));
typedef bf16  v4bf  __attribute__((ext_vector_type(4)));
typedef float v8f   __attribute__((ext_vector_type(8)));

// ---- atomics -------------------------------------------------------------
__device__ __forceinline__ void atomicAddF(float* p, float v) {
    __hip_atomic_fetch_add(p, v, __ATOMIC_RELAXED, __HIP_MEMORY_SCOPE_AGENT);
}
// float max via int/uint atomics; location pre-filled with 0xFFFFFFFF
// (int -1 loses every signed max; uint-max loses every unsigned min).
__device__ __forceinline__ void atomicMaxF(float* p, float v) {
    if (v >= 0.0f) atomicMax((int*)p, __float_as_int(v));
    else           atomicMin((unsigned int*)p, __float_as_uint(v));
}

// ---- WMMA fragment loaders (contiguous-k layouts, wave32) ------------------
// A (16x32 bf16): lane<16 -> row M=lane, K = 0..15; lane>=16 -> K+16.
// ap points at row start + akh; two contiguous 8xbf16 (16 B) loads.
__device__ __forceinline__ v16bf load_a_bf(const bf16* __restrict__ ap) {
    v8bf lo = *(const v8bf*)(ap);
    v8bf hi = *(const v8bf*)(ap + 8);
    v16bf r;
#pragma unroll
    for (int j = 0; j < 8; ++j) { r[j] = lo[j]; r[8 + j] = hi[j]; }
    return r;
}
// B (32x16 bf16) from row-major bf16 W[Hout,K]: B[k][n] = W[n][k].
// lane<16 -> col N=lane, K = k0..k0+15 ; lane>=16 -> K = k0+16..k0+31.
// 16 contiguous bf16 (32 B) load.
__device__ __forceinline__ v16bf load_b_bf(const bf16* __restrict__ wp) {
    return *(const v16bf*)(wp);
}

// ---- generic GEMM: out[M,Hout] = A[M,K] @ W[NT*16,K]^T + bias -------------
// A, W pre-converted bf16. W is padded to NT*16 rows (no B bounds checks).
template <int NT>
__global__ __launch_bounds__(256) void gemm_bf16(
    const bf16* __restrict__ A, int lda,
    const bf16* __restrict__ W, int ldw,
    const float* __restrict__ bias,
    float* __restrict__ out, int ldo, int Hout, int M, int K)
{
    int wave = (int)((blockIdx.x * blockDim.x + threadIdx.x) >> 5);
    int lane = threadIdx.x & 31;
    int m0 = wave * 16;
    if (m0 >= M) return;                      // wave-uniform: EXEC stays all-1

    int arow = lane & 15;
    int akh  = (lane >> 4) << 4;              // 0 or 16
    int bn   = lane & 15;
    int bkh  = (lane >> 4) << 4;              // 0 or 16

    const bf16* ap = A + (size_t)(m0 + arow) * lda + akh;
    v8f acc[NT] = {};
    for (int k0 = 0; k0 < K; k0 += 32) {
        v16bf af = load_a_bf(ap + k0);
#pragma unroll
        for (int t = 0; t < NT; ++t) {
            v16bf bv = load_b_bf(W + (size_t)(t * 16 + bn) * ldw + k0 + bkh);
            acc[t] = __builtin_amdgcn_wmma_f32_16x16x32_bf16(
                false, af, false, bv, (short)0, acc[t], false, false);
        }
    }
    // D layout: VGPR i -> row i (lanes 0-15) / row i+8 (lanes 16-31), col = lane&15
    int rbase = (lane >> 4) << 3;
#pragma unroll
    for (int t = 0; t < NT; ++t) {
        int n = t * 16 + (lane & 15);
        if (n >= Hout) continue;
        float bsum = bias ? bias[n] : 0.0f;
#pragma unroll
        for (int i = 0; i < 8; ++i)
            out[(size_t)(m0 + rbase + i) * ldo + n] = acc[t][i] + bsum;
    }
}

// ---- embedding: h0 = relu([maps | feats] @ Wfused^T + bfused) ------------
// All operands pre-converted bf16; feats pre-padded to K=32.
// Writes f32 h0 (edge phase / residual) and bf16 mirror (next GEMMs).
__global__ __launch_bounds__(256) void embed_wmma(
    const bf16* __restrict__ mapsB, const bf16* __restrict__ WcatMb,
    const bf16* __restrict__ featsPB, const bf16* __restrict__ W2fb,
    const float* __restrict__ bfused,
    float* __restrict__ outF, bf16* __restrict__ outB, int M)
{
    int wave = (int)((blockIdx.x * blockDim.x + threadIdx.x) >> 5);
    int lane = threadIdx.x & 31;
    int m0 = wave * 16;
    if (m0 >= M) return;

    int arow = lane & 15;
    int akh  = (lane >> 4) << 4;
    int bn   = lane & 15;
    int bkh  = (lane >> 4) << 4;

    v8f acc[8] = {};
    const bf16* ap = mapsB + (size_t)(m0 + arow) * MAPD + akh;
    for (int k0 = 0; k0 < MAPD; k0 += 32) {
        v16bf af = load_a_bf(ap + k0);
#pragma unroll
        for (int t = 0; t < 8; ++t) {
            v16bf bv = load_b_bf(WcatMb + (size_t)(t * 16 + bn) * MAPD + k0 + bkh);
            acc[t] = __builtin_amdgcn_wmma_f32_16x16x32_bf16(
                false, af, false, bv, (short)0, acc[t], false, false);
        }
    }
    {   // feats tail: pre-padded bf16 [M,32]
        v16bf af = load_a_bf(featsPB + (size_t)(m0 + arow) * 32 + akh);
#pragma unroll
        for (int t = 0; t < 8; ++t) {
            v16bf bv = load_b_bf(W2fb + (size_t)(t * 16 + bn) * 32 + bkh);
            acc[t] = __builtin_amdgcn_wmma_f32_16x16x32_bf16(
                false, af, false, bv, (short)0, acc[t], false, false);
        }
    }
    int rbase = (lane >> 4) << 3;
#pragma unroll
    for (int t = 0; t < 8; ++t) {
        int n = t * 16 + (lane & 15);
        float bsum = bfused[n];
#pragma unroll
        for (int i = 0; i < 8; ++i) {
            float v = fmaxf(acc[t][i] + bsum, 0.0f);
            size_t idx = (size_t)(m0 + rbase + i) * HDIM + n;
            outF[idx] = v;
            outB[idx] = (bf16)v;
        }
    }
}

// ---- prep: W2fb[128][32] = bf16(Wcat[:,512:] @ Weh); bfused ---------------
__global__ void prep_fuse(const float* __restrict__ Wcat, const float* __restrict__ bcat,
                          const float* __restrict__ Weh,  const float* __restrict__ beh,
                          bf16* __restrict__ W2fb, float* __restrict__ bfused)
{
    int j = threadIdx.x;                       // 0..127
    const int ldcat = MAPD + HDIM;
    for (int k = 0; k < 32; ++k) {
        float s = 0.0f;
        if (k < IN_DIM)
            for (int t = 0; t < HDIM; ++t)
                s += Wcat[(size_t)j * ldcat + MAPD + t] * Weh[(size_t)t * IN_DIM + k];
        W2fb[j * 32 + k] = (bf16)s;
    }
    float b = bcat[j];
    for (int t = 0; t < HDIM; ++t)
        b += Wcat[(size_t)j * ldcat + MAPD + t] * beh[t];
    bfused[j] = b;
}

// ---- generic strided f32 -> padded bf16 conversion -----------------------
// out[r][c] = (r<in_rows && c<cols) ? in[r*ldi+c] : 0   for r<rows, c<ldo
__global__ void cvt_rows(const float* __restrict__ in, int ldi, int in_rows, int cols,
                         bf16* __restrict__ out, int ldo, int rows)
{
    int idx = blockIdx.x * blockDim.x + threadIdx.x;
    if (idx >= rows * ldo) return;
    int r = idx / ldo, c = idx - r * ldo;
    float v = (r < in_rows && c < cols) ? in[(size_t)r * ldi + c] : 0.0f;
    out[idx] = (bf16)v;
}

__global__ void count_indeg(const int* __restrict__ dst, float* __restrict__ indeg, int E) {
    int e = blockIdx.x * blockDim.x + threadIdx.x;
    if (e < E) atomicAddF(&indeg[dst[e]], 1.0f);
}

// ---- per-node attention scores: wave per node, shuffle reduce ------------
__global__ __launch_bounds__(256) void score_kernel(
    const float* __restrict__ z, const float* __restrict__ a,
    float* __restrict__ s_src, float* __restrict__ s_dst, int M)
{
    int wave = (int)((blockIdx.x * blockDim.x + threadIdx.x) >> 5);
    int lane = threadIdx.x & 31;
    if (wave >= M) return;
    float4 zv = ((const float4*)(z + (size_t)wave * HDIM))[lane];
    float4 a0 = ((const float4*)a)[lane];
    float4 a1 = ((const float4*)(a + HDIM))[lane];
    float ss = zv.x * a0.x + zv.y * a0.y + zv.z * a0.z + zv.w * a0.w;
    float sd = zv.x * a1.x + zv.y * a1.y + zv.z * a1.z + zv.w * a1.w;
#pragma unroll
    for (int off = 16; off > 0; off >>= 1) {
        ss += __shfl_down(ss, off, 32);
        sd += __shfl_down(sd, off, 32);
    }
    if (lane == 0) { s_src[wave] = ss; s_dst[wave] = sd; }
}

// ---- edge phase ----------------------------------------------------------
__global__ void edge_max_kernel(const int* __restrict__ src, const int* __restrict__ dst,
                                const float* __restrict__ s_src, const float* __restrict__ s_dst,
                                float* __restrict__ ew, float* __restrict__ m, int E)
{
    int e = blockIdx.x * blockDim.x + threadIdx.x;
    if (e >= E) return;
    float v = s_src[src[e]] + s_dst[dst[e]];
    v = v > 0.0f ? v : 0.01f * v;              // leaky_relu
    ew[e] = v;
    atomicMaxF(&m[dst[e]], v);
}

__global__ void edge_expsum_kernel(const int* __restrict__ dst, float* __restrict__ ew,
                                   const float* __restrict__ m, float* __restrict__ denom, int E)
{
    int e = blockIdx.x * blockDim.x + threadIdx.x;
    if (e >= E) return;
    int d = dst[e];
    float ex = __expf(ew[e] - m[d]);
    ew[e] = ex;
    atomicAddF(&denom[d], ex);
}

// wave per edge: gather 128-wide z[src], scatter-add alpha*z into agg[dst]
__global__ __launch_bounds__(256) void agg_kernel(
    const int* __restrict__ src, const int* __restrict__ dst,
    const float* __restrict__ ew, const float* __restrict__ denom,
    const float* __restrict__ z, float* __restrict__ agg, int E)
{
    int wave = (int)((blockIdx.x * blockDim.x + threadIdx.x) >> 5);
    int lane = threadIdx.x & 31;
    if (wave >= E) return;
    int s = src[wave], d = dst[wave];
    float alpha = ew[wave] / fmaxf(denom[d], 1e-16f);
    float4 zv = ((const float4*)(z + (size_t)s * HDIM))[lane];
    float* ag = agg + (size_t)d * HDIM + lane * 4;
    atomicAddF(ag + 0, alpha * zv.x);
    atomicAddF(ag + 1, alpha * zv.y);
    atomicAddF(ag + 2, alpha * zv.z);
    atomicAddF(ag + 3, alpha * zv.w);
}

// h_out = indeg>0 ? h_in + h_s + agg : 2*h_in ; writes f32 + bf16 mirror
__global__ void update_kernel(const float* __restrict__ h_in, const float* __restrict__ hs,
                              const float* __restrict__ agg, const float* __restrict__ indeg,
                              float* __restrict__ h_outF, bf16* __restrict__ h_outB, int M)
{
    int i = blockIdx.x * blockDim.x + threadIdx.x;   // over M*32 float4 chunks
    if (i >= M * (HDIM / 4)) return;
    int n = i >> 5;
    float4 hi = ((const float4*)h_in)[i];
    float4 r;
    if (indeg[n] > 0.0f) {
        float4 a = ((const float4*)hs)[i];
        float4 b = ((const float4*)agg)[i];
        r = make_float4(hi.x + a.x + b.x, hi.y + a.y + b.y,
                        hi.z + a.z + b.z, hi.w + a.w + b.w);
    } else {
        r = make_float4(2.0f * hi.x, 2.0f * hi.y, 2.0f * hi.z, 2.0f * hi.w);
    }
    ((float4*)h_outF)[i] = r;
    v4bf rb;
    rb[0] = (bf16)r.x; rb[1] = (bf16)r.y; rb[2] = (bf16)r.z; rb[3] = (bf16)r.w;
    ((v4bf*)h_outB)[i] = rb;
}

// ---------------------------------------------------------------------------
extern "C" void kernel_launch(void* const* d_in, const int* in_sizes, int n_in,
                              void* d_out, int out_size, void* d_ws, size_t ws_size,
                              hipStream_t stream)
{
    const float* feats = (const float*)d_in[0];
    const float* maps  = (const float*)d_in[2];
    const int*   src   = (const int*)d_in[4];
    const int*   dst   = (const int*)d_in[5];
    const float* Weh   = (const float*)d_in[6];
    const float* beh   = (const float*)d_in[7];
    const float* Wcat  = (const float*)d_in[10];
    const float* bcat  = (const float*)d_in[11];
    const float* Wlay[2][2] = {{(const float*)d_in[12], (const float*)d_in[13]},
                               {(const float*)d_in[15], (const float*)d_in[16]}};
    const float* alay[2] = {(const float*)d_in[14], (const float*)d_in[17]};
    const float* Wout = (const float*)d_in[18];
    const float* bout = (const float*)d_in[19];
    float* out = (float*)d_out;

    // ---- workspace carve-up: f32 region, then bf16 region ----------------
    // All offsets kept multiples of 8 floats (32 B) for vector-load alignment.
    float* ws = (float*)d_ws;
    size_t off = 0;
    const size_t NH = (size_t)N_NODES * HDIM;
    float* hA    = ws + off; off += NH;
    float* hB    = ws + off; off += NH;
    float* hs    = ws + off; off += NH;
    float* zb    = ws + off; off += NH;
    float* agg   = ws + off; off += NH;
    float* m     = ws + off; off += N_NODES;
    float* denom = ws + off; off += N_NODES;
    float* indeg = ws + off; off += N_NODES;
    float* ssrc  = ws + off; off += N_NODES;
    float* sdst  = ws + off; off += N_NODES;
    float* ew    = ws + off; off += N_EDGES;
    float* bfus  = ws + off; off += 128;

    bf16* bws = (bf16*)(ws + off);
    size_t boff = 0;
    bf16* mapsB   = bws + boff; boff += (size_t)N_NODES * MAPD;
    bf16* featsPB = bws + boff; boff += (size_t)N_NODES * 32;
    bf16* hAb     = bws + boff; boff += NH;
    bf16* hBb     = bws + boff; boff += NH;
    bf16* WcatMb  = bws + boff; boff += (size_t)HDIM * MAPD;
    bf16* W2fb    = bws + boff; boff += (size_t)HDIM * 32;
    bf16* Wb[2][2];
    for (int l = 0; l < 2; ++l)
        for (int k = 0; k < 2; ++k) { Wb[l][k] = bws + boff; boff += (size_t)HDIM * HDIM; }
    bf16* Woutb   = bws + boff; boff += (size_t)32 * HDIM;   // padded 24 -> 32 rows

    const int WPB = 8;                               // waves per 256-thread block
    int mtiles = (N_NODES + 15) / 16;                // 1250
    int gemm_blocks = (mtiles + WPB - 1) / WPB;      // 157
    int edge_blocks = (N_EDGES + 255) / 256;         // 2500
    int node_wave_blocks = (N_NODES + WPB - 1) / WPB;
    int agg_blocks = (N_EDGES + WPB - 1) / WPB;      // 80000

    // ---- one-time (per launch) conversions & degree count ----------------
    prep_fuse<<<1, 128, 0, stream>>>(Wcat, bcat, Weh, beh, W2fb, bfus);
    hipMemsetAsync(indeg, 0, N_NODES * sizeof(float), stream);
    count_indeg<<<edge_blocks, 256, 0, stream>>>(dst, indeg, N_EDGES);

    {   // bf16 conversions
        int n;
        n = N_NODES * MAPD;
        cvt_rows<<<(n + 255) / 256, 256, 0, stream>>>(maps, MAPD, N_NODES, MAPD, mapsB, MAPD, N_NODES);
        n = N_NODES * 32;
        cvt_rows<<<(n + 255) / 256, 256, 0, stream>>>(feats, IN_DIM, N_NODES, IN_DIM, featsPB, 32, N_NODES);
        n = HDIM * MAPD;
        cvt_rows<<<(n + 255) / 256, 256, 0, stream>>>(Wcat, MAPD + HDIM, HDIM, MAPD, WcatMb, MAPD, HDIM);
        n = HDIM * HDIM;
        for (int l = 0; l < 2; ++l)
            for (int k = 0; k < 2; ++k)
                cvt_rows<<<(n + 255) / 256, 256, 0, stream>>>(Wlay[l][k], HDIM, HDIM, HDIM,
                                                              Wb[l][k], HDIM, HDIM);
        n = 32 * HDIM;
        cvt_rows<<<(n + 255) / 256, 256, 0, stream>>>(Wout, HDIM, OUTD, HDIM, Woutb, HDIM, 32);
    }

    // ---- embedding --------------------------------------------------------
    embed_wmma<<<gemm_blocks, 256, 0, stream>>>(mapsB, WcatMb, featsPB, W2fb, bfus,
                                                hA, hAb, N_NODES);

    // ---- two GAT layers ---------------------------------------------------
    float* hinF = hA;  bf16* hinB = hAb;
    float* houtF = hB; bf16* houtB = hBb;
    for (int l = 0; l < 2; ++l) {
        gemm_bf16<8><<<gemm_blocks, 256, 0, stream>>>(hinB, HDIM, Wb[l][0], HDIM,
                                                      nullptr, hs, HDIM, HDIM, N_NODES, HDIM);
        gemm_bf16<8><<<gemm_blocks, 256, 0, stream>>>(hinB, HDIM, Wb[l][1], HDIM,
                                                      nullptr, zb, HDIM, HDIM, N_NODES, HDIM);
        score_kernel<<<node_wave_blocks, 256, 0, stream>>>(zb, alay[l], ssrc, sdst, N_NODES);
        hipMemsetAsync(m, 0xFF, N_NODES * sizeof(float), stream);   // -NaN sentinel, any value wins
        hipMemsetAsync(denom, 0, N_NODES * sizeof(float), stream);
        hipMemsetAsync(agg, 0, NH * sizeof(float), stream);
        edge_max_kernel<<<edge_blocks, 256, 0, stream>>>(src, dst, ssrc, sdst, ew, m, N_EDGES);
        edge_expsum_kernel<<<edge_blocks, 256, 0, stream>>>(dst, ew, m, denom, N_EDGES);
        agg_kernel<<<agg_blocks, 256, 0, stream>>>(src, dst, ew, denom, zb, agg, N_EDGES);
        update_kernel<<<(N_NODES * (HDIM / 4) + 255) / 256, 256, 0, stream>>>(
            hinF, hs, agg, indeg, houtF, houtB, N_NODES);
        float* tf = hinF; hinF = houtF; houtF = tf;
        bf16*  tb = hinB; hinB = houtB; houtB = tb;
    }

    // ---- output projection (Hout=24, weights padded to 32 rows) ----------
    gemm_bf16<2><<<gemm_blocks, 256, 0, stream>>>(hinB, HDIM, Woutb, HDIM,
                                                  bout, out, OUTD, OUTD, N_NODES, HDIM);
}
